// CausalAttention_66297115181556
// MI455X (gfx1250) — matrix operand order repaired and checked by reading, not compiled, not run on hardware
//
#include <hip/hip_runtime.h>
#include <hip/hip_bf16.h>

typedef __attribute__((ext_vector_type(16))) _Float16 v16h;
typedef __attribute__((ext_vector_type(8)))  _Float16 v8h;
typedef __attribute__((ext_vector_type(8)))  float    v8f;

#define DIMC   512
#define NSEQ   2048
#define BATCH  2
#define HEADS  8
#define DHEAD  64
#define ROWS   (BATCH * NSEQ)          // 4096

// ---------------------------------------------------------------------------
// LayerNorm over last dim (512), fp32 in -> f16 out
// ---------------------------------------------------------------------------
__global__ __launch_bounds__(256) void ln_kernel(const float* __restrict__ x,
                                                 const float* __restrict__ gamma,
                                                 const float* __restrict__ beta,
                                                 _Float16* __restrict__ xn) {
  __shared__ float red[16];
  const int row = blockIdx.x;
  const float* xr = x + (size_t)row * DIMC;
  const int t = threadIdx.x;
  float v0 = xr[t];
  float v1 = xr[t + 256];
  float s  = v0 + v1;
  float sq = v0 * v0 + v1 * v1;
  for (int off = 1; off < 32; off <<= 1) {
    s  += __shfl_xor(s,  off, 32);
    sq += __shfl_xor(sq, off, 32);
  }
  const int wave = t >> 5, lane = t & 31;
  if (lane == 0) { red[wave] = s; red[8 + wave] = sq; }
  __syncthreads();
  float ts = 0.f, tq = 0.f;
  for (int w = 0; w < 8; ++w) { ts += red[w]; tq += red[8 + w]; }
  const float mean = ts * (1.0f / DIMC);
  const float var  = tq * (1.0f / DIMC) - mean * mean;
  const float rstd = rsqrtf(var + 1e-5f);
  _Float16* xo = xn + (size_t)row * DIMC;
  xo[t]       = (_Float16)((v0 - mean) * rstd * gamma[t]       + beta[t]);
  xo[t + 256] = (_Float16)((v1 - mean) * rstd * gamma[t + 256] + beta[t + 256]);
}

// ---------------------------------------------------------------------------
// fp32 -> f16 convert
// ---------------------------------------------------------------------------
__global__ void cvt_kernel(const float* __restrict__ src, _Float16* __restrict__ dst, int n) {
  int i = blockIdx.x * 256 + threadIdx.x;
  if (i < n) dst[i] = (_Float16)src[i];
}

// ---------------------------------------------------------------------------
// C[r, o] = sum_d A[r, d] * W[o, d]   (A: [4096,512] f16, W: [512,512] f16)
// Block tile 128x64, 8 waves each doing 32x32 via 2x2 WMMA frags.
// ---------------------------------------------------------------------------
__global__ __launch_bounds__(256) void gemm_f16(const _Float16* __restrict__ A,
                                                const _Float16* __restrict__ W,
                                                float* __restrict__ C) {
  const int lane = threadIdx.x & 31;
  const int wave = threadIdx.x >> 5;
  const int wm = wave & 3, wn = wave >> 2;
  const int row0 = blockIdx.y * 128 + wm * 32;
  const int col0 = blockIdx.x * 64  + wn * 32;
  const int lm = lane & 15, half = lane >> 4;

  v8f c[2][2] = {};
  for (int k0 = 0; k0 < DIMC; k0 += 32) {
    v16h a[2], b[2];
#pragma unroll
    for (int i = 0; i < 2; ++i) {
      const _Float16* ap = A + (size_t)(row0 + i * 16 + lm) * DIMC + k0;
      v8h alo = *(const v8h*)(ap + 16 * half);
      v8h ahi = *(const v8h*)(ap + 16 * half + 8);
      const _Float16* bp = W + (size_t)(col0 + i * 16 + lm) * DIMC + k0 + 16 * half;
      v8h blo = *(const v8h*)(bp);
      v8h bhi = *(const v8h*)(bp + 8);
#pragma unroll
      for (int e = 0; e < 8; ++e) {
        a[i][e] = alo[e]; a[i][e + 8] = ahi[e];
        b[i][e] = blo[e]; b[i][e + 8] = bhi[e];
      }
    }
#pragma unroll
    for (int i = 0; i < 2; ++i)
#pragma unroll
      for (int j = 0; j < 2; ++j)
        c[i][j] = __builtin_amdgcn_wmma_f32_16x16x32_f16(
            false, a[i], false, b[j], (short)0, c[i][j], false, false);
  }
#pragma unroll
  for (int i = 0; i < 2; ++i)
#pragma unroll
    for (int j = 0; j < 2; ++j)
#pragma unroll
      for (int r = 0; r < 8; ++r) {
        int m = row0 + i * 16 + r + 8 * half;
        int n = col0 + j * 16 + lm;
        C[(size_t)m * DIMC + n] = c[i][j][r];
      }
}

// ---------------------------------------------------------------------------
// Depthwise causal conv (k=3) epilogue + head-split relayout -> [b,h,n,64] f16
// ---------------------------------------------------------------------------
__global__ __launch_bounds__(256) void dwconv_kernel(const float* __restrict__ Y,
                                                     const float* __restrict__ wd,
                                                     _Float16* __restrict__ out,
                                                     float scale) {
  const int idx = blockIdx.x * 256 + threadIdx.x;   // b*N*512 + n*512 + e
  const int e = idx & 511;
  const int n = (idx >> 9) & 2047;
  const int b = idx >> 20;
  const float* yp = Y + (size_t)idx;
  float v = wd[e * 3 + 2] * yp[0];
  if (n >= 1) v += wd[e * 3 + 1] * yp[-DIMC];
  if (n >= 2) v += wd[e * 3 + 0] * yp[-2 * DIMC];
  v *= scale;
  const int h = e >> 6, d = e & 63;
  out[(((size_t)(b * HEADS + h)) * NSEQ + n) * DHEAD + d] = (_Float16)v;
}

// ---------------------------------------------------------------------------
// Flash attention: 1 block = (b,h, 64-query-row tile), 4 waves x 16 rows.
// Causal + ALiBi, online softmax, all matmuls via v_wmma_f32_16x16x32_f16.
// ---------------------------------------------------------------------------
__global__ __launch_bounds__(128) void attn_kernel(const _Float16* __restrict__ Q,
                                                   const _Float16* __restrict__ K,
                                                   const _Float16* __restrict__ V,
                                                   _Float16* __restrict__ O) {
  __shared__ __align__(16) _Float16 vt[64 * 72];        // V^T tile, padded rows
  __shared__ __align__(16) _Float16 pbuf[4][16 * 64];   // per-wave P staging

  const int bh = blockIdx.x >> 5;   // 32 query tiles per (b,h)
  const int qi = blockIdx.x & 31;
  const int b  = bh >> 3, h = bh & 7;
  const int lane = threadIdx.x & 31, wave = threadIdx.x >> 5;
  const int lm = lane & 15, half = lane >> 4;
  const float slope = exp2f(-(float)(h + 1));

  const _Float16* Qb = Q + (size_t)bh * NSEQ * DHEAD;
  const _Float16* Kb = K + (size_t)bh * NSEQ * DHEAD;
  const _Float16* Vb = V + (size_t)bh * NSEQ * DHEAD;
  const int i0 = qi * 64 + wave * 16;   // first query row of this wave

  // Q fragments (A-matrix layout), held for whole kernel. q already scaled.
  v16h qa[2];
  {
    const _Float16* qp = Qb + (size_t)(i0 + lm) * DHEAD;
#pragma unroll
    for (int ks = 0; ks < 2; ++ks) {
      v8h lo = *(const v8h*)(qp + ks * 32 + 16 * half);
      v8h hi = *(const v8h*)(qp + ks * 32 + 16 * half + 8);
#pragma unroll
      for (int e = 0; e < 8; ++e) { qa[ks][e] = lo[e]; qa[ks][e + 8] = hi[e]; }
    }
  }

  float mstate[8], lstate[8];
  v8f oacc[4] = {};
#pragma unroll
  for (int r = 0; r < 8; ++r) { mstate[r] = -3.4e38f; lstate[r] = 0.f; }

  const int njt = qi + 1;   // uniform trip count across the 4 waves
  for (int jt = 0; jt < njt; ++jt) {
    const int j0 = jt * 64;

    // ---- stage V^T tile into LDS (cooperative) ----
    __syncthreads();
    for (int it = 0; it < 32; ++it) {
      const int idx = it * 128 + threadIdx.x;
      const int jj = idx >> 6, dd = idx & 63;
      vt[dd * 72 + jj] = Vb[(size_t)(j0 + jj) * DHEAD + dd];
    }
    __syncthreads();

    if (jt + 1 < njt)   // pull next K tile toward L2/L0
      __builtin_prefetch(Kb + (size_t)(j0 + 64 + lane) * DHEAD, 0, 0);

    // ---- S = Q K^T (16 x 64), K rows load directly as B fragments ----
    v8f s[4];
#pragma unroll
    for (int nf = 0; nf < 4; ++nf) {
      v8f acc = {};
#pragma unroll
      for (int ks = 0; ks < 2; ++ks) {
        const _Float16* kp = Kb + (size_t)(j0 + nf * 16 + lm) * DHEAD + ks * 32 + 16 * half;
        v8h blo = *(const v8h*)kp;
        v8h bhi = *(const v8h*)(kp + 8);
        v16h kb;
#pragma unroll
        for (int e = 0; e < 8; ++e) { kb[e] = blo[e]; kb[e + 8] = bhi[e]; }
        acc = __builtin_amdgcn_wmma_f32_16x16x32_f16(
            false, qa[ks], false, kb, (short)0, acc, false, false);
      }
      s[nf] = acc;
    }

    // ---- ALiBi + causal mask + online softmax ----
    float mnew[8], corr[8];
#pragma unroll
    for (int r = 0; r < 8; ++r) {
      const int i = i0 + r + 8 * half;
      float mx = -3.4e38f;
#pragma unroll
      for (int nf = 0; nf < 4; ++nf) {
        const int j = j0 + nf * 16 + lm;
        float sv = s[nf][r] - slope * fabsf((float)(i - j));
        if (j > i) sv = -1e30f;
        s[nf][r] = sv;
        mx = fmaxf(mx, sv);
      }
      for (int off = 1; off < 16; off <<= 1) mx = fmaxf(mx, __shfl_xor(mx, off, 32));
      mnew[r] = fmaxf(mstate[r], mx);
      corr[r] = __expf(mstate[r] - mnew[r]);
      mstate[r] = mnew[r];
    }

    _Float16* pw = &pbuf[wave][0];
#pragma unroll
    for (int r = 0; r < 8; ++r) {
      float rs = 0.f;
#pragma unroll
      for (int nf = 0; nf < 4; ++nf) {
        const float pv = __expf(s[nf][r] - mnew[r]);
        rs += pv;
        pw[(r + 8 * half) * 64 + nf * 16 + lm] = (_Float16)pv;
      }
      for (int off = 1; off < 16; off <<= 1) rs += __shfl_xor(rs, off, 32);
      lstate[r] = lstate[r] * corr[r] + rs;
#pragma unroll
      for (int nf2 = 0; nf2 < 4; ++nf2) oacc[nf2][r] *= corr[r];
    }

    // ---- O += P V : A-frags from per-wave LDS, B-frags from V^T LDS ----
#pragma unroll
    for (int ks = 0; ks < 2; ++ks) {
      v16h pa;
      const _Float16* pp = pw + (size_t)lm * 64 + ks * 32;
      v8h plo = *(const v8h*)(pp + 16 * half);
      v8h phi = *(const v8h*)(pp + 16 * half + 8);
#pragma unroll
      for (int e = 0; e < 8; ++e) { pa[e] = plo[e]; pa[e + 8] = phi[e]; }
#pragma unroll
      for (int nf2 = 0; nf2 < 4; ++nf2) {
        const _Float16* vp = &vt[(nf2 * 16 + lm) * 72 + ks * 32 + 16 * half];
        v8h vlo = *(const v8h*)vp;
        v8h vhi = *(const v8h*)(vp + 8);
        v16h vb;
#pragma unroll
        for (int e = 0; e < 8; ++e) { vb[e] = vlo[e]; vb[e + 8] = vhi[e]; }
        oacc[nf2] = __builtin_amdgcn_wmma_f32_16x16x32_f16(
            false, pa, false, vb, (short)0, oacc[nf2], false, false);
      }
    }
  }

  // ---- normalize + merge heads -> [b*N, 512] f16 ----
#pragma unroll
  for (int r = 0; r < 8; ++r) {
    const float inv = 1.0f / lstate[r];
    const int i = i0 + r + 8 * half;
    const size_t rowbase = ((size_t)(b * NSEQ + i)) * DIMC + h * DHEAD;
#pragma unroll
    for (int nf2 = 0; nf2 < 4; ++nf2)
      O[rowbase + nf2 * 16 + lm] = (_Float16)(oacc[nf2][r] * inv);
  }
}

// ---------------------------------------------------------------------------
extern "C" void kernel_launch(void* const* d_in, const int* in_sizes, int n_in,
                              void* d_out, int out_size, void* d_ws, size_t ws_size,
                              hipStream_t stream) {
  const float* x     = (const float*)d_in[0];
  const float* gamma = (const float*)d_in[1];
  const float* beta  = (const float*)d_in[2];
  const float* wq1   = (const float*)d_in[3];
  const float* wqd   = (const float*)d_in[4];
  const float* wk1   = (const float*)d_in[5];
  const float* wkd   = (const float*)d_in[6];
  const float* wv1   = (const float*)d_in[7];
  const float* wvd   = (const float*)d_in[8];
  const float* wout  = (const float*)d_in[9];

  // workspace layout (~30 MB)
  _Float16* xn   = (_Float16*)d_ws;             // 4096*512
  _Float16* wqh  = xn  + (size_t)ROWS * DIMC;   // 512*512 each
  _Float16* wkh  = wqh + (size_t)DIMC * DIMC;
  _Float16* wvh  = wkh + (size_t)DIMC * DIMC;
  _Float16* woh  = wvh + (size_t)DIMC * DIMC;
  float*    ybuf = (float*)(woh + (size_t)DIMC * DIMC);   // 4096*512 fp32
  _Float16* qh   = (_Float16*)(ybuf + (size_t)ROWS * DIMC);
  _Float16* kh   = qh + (size_t)ROWS * DIMC;
  _Float16* vh   = kh + (size_t)ROWS * DIMC;
  _Float16* atn  = vh + (size_t)ROWS * DIMC;

  // 1) LayerNorm -> f16
  ln_kernel<<<ROWS, 256, 0, stream>>>(x, gamma, beta, xn);

  // 2) weights -> f16
  const int wn = DIMC * DIMC;
  cvt_kernel<<<(wn + 255) / 256, 256, 0, stream>>>(wq1,  wqh, wn);
  cvt_kernel<<<(wn + 255) / 256, 256, 0, stream>>>(wk1,  wkh, wn);
  cvt_kernel<<<(wn + 255) / 256, 256, 0, stream>>>(wv1,  wvh, wn);
  cvt_kernel<<<(wn + 255) / 256, 256, 0, stream>>>(wout, woh, wn);

  dim3 ggrid(DIMC / 64, ROWS / 128);
  const int cvblocks = (ROWS * DIMC) / 256;

  // 3) Q/K/V: pointwise GEMM (WMMA) + depthwise causal conv + head split
  gemm_f16<<<ggrid, 256, 0, stream>>>(xn, wqh, ybuf);
  dwconv_kernel<<<cvblocks, 256, 0, stream>>>(ybuf, wqd, qh, 0.125f);  // * DH^-0.5
  gemm_f16<<<ggrid, 256, 0, stream>>>(xn, wkh, ybuf);
  dwconv_kernel<<<cvblocks, 256, 0, stream>>>(ybuf, wkd, kh, 1.0f);
  gemm_f16<<<ggrid, 256, 0, stream>>>(xn, wvh, ybuf);
  dwconv_kernel<<<cvblocks, 256, 0, stream>>>(ybuf, wvd, vh, 1.0f);

  // 4) flash attention: B*H*(N/64) = 512 blocks of 128 threads
  attn_kernel<<<BATCH * HEADS * (NSEQ / 64), 128, 0, stream>>>(qh, kh, vh, atn);

  // 5) output projection -> fp32 d_out
  gemm_f16<<<ggrid, 256, 0, stream>>>(atn, woh, (float*)d_out);
}
